// MixedChunkAttention_80247168959186
// MI455X (gfx1250) — matrix operand order repaired and checked by reading, not compiled, not run on hardware
//
#include <hip/hip_runtime.h>
#include <hip/hip_bf16.h>

// ---------------------------------------------------------------------------
// MixedChunkAttention for MI455X (gfx1250)
//  - every GEMM on v_wmma_f32_16x16x32_f16 (f32 accumulate)
//  - all B operands pre-transposed to [N][K] so LDS fills are pure b128 tiles
//  - heavy GEMMs stream tiles with the Tensor Data Mover (tensor_load_to_lds,
//    double-buffered, s_wait_tensorcnt), using TDM LDS padding for bank-safe
//    stride-40 rows
// ---------------------------------------------------------------------------

typedef _Float16 f16;
typedef __attribute__((ext_vector_type(8)))  _Float16 v8h;
typedef __attribute__((ext_vector_type(16))) _Float16 v16h;
typedef __attribute__((ext_vector_type(8)))  float    v8f;
typedef __attribute__((ext_vector_type(4)))  unsigned int v4u;
typedef __attribute__((ext_vector_type(8)))  int      v8i32;
typedef __attribute__((ext_vector_type(4)))  int      v4i32;

union F16x16 { v16h v; v8h h[2]; };

#define NB      4
#define SEQ     4096
#define DIM     1024
#define EXP     2048     // INNER
#define HEAD_D  128
#define CHUNK   256      // GROUP
#define NCHUNK  (SEQ / CHUNK)          // 16
#define ROWS    (NB * SEQ)             // 16384
#define NCHUNKS_TOT (NB * NCHUNK)      // 64

#define INV_SCALE 0.02209708691207961f   // 1/sqrt(2048)

#define LDS_STRIDE 40                    // 32 + 8 halfword pad (80B rows)
#define TILE_F16   (128 * LDS_STRIDE)

// ---------------------------------------------------------------------------
// Plain tile fill: dst[128][LDS_STRIDE] <- src[128][0..31], row-major, b128 only
// ---------------------------------------------------------------------------
__device__ inline void load_tile(f16* __restrict__ dst, const f16* __restrict__ src, int ld) {
    int t   = threadIdx.x;
    int r   = t >> 1;
    int seg = (t & 1) * 16;
    const v8h* s = (const v8h*)(src + (size_t)r * ld + seg);
    v8h a = s[0];
    v8h b = s[1];
    v8h* d = (v8h*)(dst + r * LDS_STRIDE + seg);
    d[0] = a; d[1] = b;
}

__device__ inline void prefetch_tile(const f16* src, int ld) {
    int t = threadIdx.x;
    __builtin_prefetch(src + (size_t)(t >> 1) * ld + (t & 1) * 16, 0, 0);
}

// ---------------------------------------------------------------------------
// Tensor Data Mover: DMA one 128x32 f16 tile (row stride ld elems) into LDS
// with 4-DWORD padding after every 16 DWORDs => LDS row pitch = 80B = stride 40.
// D# layout per CDNA5 ISA ch.8 (group0: 128b, group1: 256b, groups2/3 unused).
// This toolchain exposes the 6-arg builtin:
//   (uint32x4 g0, int32x8 g1, int32x4 g2, int32x4 g3, int32x8, i32 cpol)
// ---------------------------------------------------------------------------
__device__ inline void tdm_load_tile(const f16* src, int ld, f16* lds_dst) {
    unsigned long long ga = (unsigned long long)(uintptr_t)src;
    unsigned int lo = (unsigned int)(uintptr_t)lds_dst;   // low 32b = LDS offset

    v4u g0;
    g0[0] = 1u;                                    // count=1, user descriptor
    g0[1] = lo;                                    // lds_addr [63:32]
    g0[2] = (unsigned int)ga;                      // global_addr[31:0]  (bits 95:64)
    g0[3] = (unsigned int)(ga >> 32) | (2u << 30); // global_addr[56:32] + type=2

    unsigned int uld = (unsigned int)ld;
    v8i32 g1;
    g1[0] = (int)((1u << 16)        // data_size = 2 bytes
                | (1u << 20)        // pad_enable
                | (3u << 22)        // pad_interval: 16 DWORDs
                | (3u << 25));      // pad_amount  : 4 DWORDs
    g1[1] = (int)((uld & 0xFFFFu) << 16);            // tensor_dim0 [63:48]
    g1[2] = (int)((uld >> 16) | (0xFFFFu << 16));    // tensor_dim0 hi, tensor_dim1 lo
    g1[3] = (int)(0xFu | (32u << 16));               // tensor_dim1 hi (=0xFFFFF), tile_dim0=32
    g1[4] = 128;                                     // tile_dim1=128, tile_dim2=0
    g1[5] = (int)uld;                                // tensor_dim0_stride lo
    g1[6] = 0;                                       // stride hi / dim1_stride lo
    g1[7] = 0;
    v4i32 z4 = {0, 0, 0, 0};
    v8i32 z8 = {0, 0, 0, 0, 0, 0, 0, 0};
    __builtin_amdgcn_tensor_load_to_lds(g0, g1, z4, z4, z8, 0);
}

// ---------------------------------------------------------------------------
// WMMA fragments (ISA 7.12.2, wave32)
// ---------------------------------------------------------------------------
__device__ inline v16h frag_a(const f16* lds, int row) {
    int lane = threadIdx.x & 31;
    int m    = row + (lane & 15);
    int kh   = (lane >> 4) * 8;
    F16x16 f;
    f.h[0] = *(const v8h*)(lds + m * LDS_STRIDE + kh);
    f.h[1] = *(const v8h*)(lds + m * LDS_STRIDE + 16 + kh);
    return f.v;
}

__device__ inline v16h frag_b(const f16* ldsBt, int col) {
    int lane = threadIdx.x & 31;
    int n    = col + (lane & 15);
    int kh   = (lane >> 4) * 8;
    F16x16 f;
    f.h[0] = *(const v8h*)(ldsBt + n * LDS_STRIDE + kh);
    f.h[1] = *(const v8h*)(ldsBt + n * LDS_STRIDE + kh + 16);
    return f.v;
}

__device__ inline void mma_tiles(const f16* ldsA, const f16* ldsB, v8f (&acc)[2][4]) {
    int w  = threadIdx.x >> 5;
    int wr = (w & 3) * 32;
    int wc = (w >> 2) * 64;
    v16h a0 = frag_a(ldsA, wr);
    v16h a1 = frag_a(ldsA, wr + 16);
#pragma unroll
    for (int c = 0; c < 4; ++c) {
        v16h b = frag_b(ldsB, wc + c * 16);
        acc[0][c] = __builtin_amdgcn_wmma_f32_16x16x32_f16(false, a0, false, b,
                        (short)0, acc[0][c], false, false);
        acc[1][c] = __builtin_amdgcn_wmma_f32_16x16x32_f16(false, a1, false, b,
                        (short)0, acc[1][c], false, false);
    }
}

// Vanilla K-loop: A [M,K] row-major, B [N,K] row-major (pre-transposed).
__device__ inline void gemm_Kloop(const f16* __restrict__ A, int lda,
                                  const f16* __restrict__ B, int ldb, int K,
                                  f16* ldsA, f16* ldsB, v8f (&acc)[2][4]) {
    for (int k0 = 0; k0 < K; k0 += 32) {
        load_tile(ldsA, A + k0, lda);
        load_tile(ldsB, B + k0, ldb);
        if (k0 + 32 < K) {
            prefetch_tile(A + k0 + 32, lda);
            prefetch_tile(B + k0 + 32, ldb);
        }
        __syncthreads();
        mma_tiles(ldsA, ldsB, acc);
        __syncthreads();
    }
}

// TDM K-loop: double-buffered tensor_load_to_lds, wave 0 issues + waits.
__device__ inline void gemm_Kloop_tdm(const f16* __restrict__ A, int lda,
                                      const f16* __restrict__ B, int ldb, int K,
                                      f16* ldsA0, f16* ldsB0, f16* ldsA1, f16* ldsB1,
                                      v8f (&acc)[2][4]) {
    bool issuer = (threadIdx.x < 32);
    if (issuer) {
        tdm_load_tile(A, lda, ldsA0);
        tdm_load_tile(B, ldb, ldsB0);
    }
    int nsteps = K >> 5;
    for (int i = 0; i < nsteps; ++i) {
        const f16* lA = (i & 1) ? ldsA1 : ldsA0;
        const f16* lB = (i & 1) ? ldsB1 : ldsB0;
        if (issuer) {
            if (i + 1 < nsteps) {
                f16* nA = (i & 1) ? ldsA0 : ldsA1;
                f16* nB = (i & 1) ? ldsB0 : ldsB1;
                tdm_load_tile(A + (size_t)(i + 1) * 32, lda, nA);
                tdm_load_tile(B + (size_t)(i + 1) * 32, ldb, nB);
                __builtin_amdgcn_s_wait_tensorcnt((short)2);  // current pair landed
            } else {
                __builtin_amdgcn_s_wait_tensorcnt((short)0);
            }
        }
        __syncthreads();              // publish TDM data to all waves
        mma_tiles(lA, lB, acc);
        __syncthreads();              // protect buffer before next overwrite
    }
}

__device__ inline float silu_f(float v) { return v / (1.f + __expf(-v)); }

// ---------------------------------------------------------------------------
// Kernels
// ---------------------------------------------------------------------------
__global__ void k_convert(const float* __restrict__ s, f16* __restrict__ d, int n) {
    int i      = blockIdx.x * blockDim.x + threadIdx.x;
    int stride = gridDim.x * blockDim.x;
    for (; i < n; i += stride) d[i] = (f16)s[i];
}

// src[K][N] fp32  ->  dst[N][K] f16  (weight transpose-convert, one-off)
__global__ void k_convT(const float* __restrict__ src, f16* __restrict__ dst, int K, int N) {
    size_t i      = (size_t)blockIdx.x * blockDim.x + threadIdx.x;
    size_t total  = (size_t)K * N;
    size_t stride = (size_t)gridDim.x * blockDim.x;
    for (; i < total; i += stride) {
        int k = (int)(i / N), n = (int)(i % N);
        dst[(size_t)n * K + k] = (f16)src[i];
    }
}

// out = silu(A @ W^T + bias).  A:[ROWS,K], W:[N,K].  grid=(ROWS/128, N/128)
// mode 0: row-major out[m*N+n];  mode 1: chunk-transposed out[(chunk*N+n)*CHUNK + m%CHUNK]
__global__ void __launch_bounds__(256)
k_gemm_silu(const f16* __restrict__ A, const f16* __restrict__ W,
            const float* __restrict__ bias, f16* __restrict__ out,
            int N, int K, int mode) {
    __shared__ f16 ldsA0[TILE_F16], ldsB0[TILE_F16], ldsA1[TILE_F16], ldsB1[TILE_F16];
    int m0 = blockIdx.x * 128, n0 = blockIdx.y * 128;
    v8f acc[2][4] = {};
    gemm_Kloop_tdm(A + (size_t)m0 * K, K, W + (size_t)n0 * K, K, K,
                   ldsA0, ldsB0, ldsA1, ldsB1, acc);

    int lane = threadIdx.x & 31, w = threadIdx.x >> 5;
    int wr = (w & 3) * 32, wc = (w >> 2) * 64;
#pragma unroll
    for (int r = 0; r < 2; ++r)
#pragma unroll
        for (int c = 0; c < 4; ++c) {
            int   n  = n0 + wc + c * 16 + (lane & 15);
            float bn = bias[n];
#pragma unroll
            for (int j = 0; j < 8; ++j) {
                int   m = m0 + wr + r * 16 + (lane >> 4) * 8 + j;
                float v = silu_f(acc[r][c][j] + bn);
                if (mode == 0) {
                    out[(size_t)m * N + n] = (f16)v;
                } else {
                    out[((size_t)(m >> 8) * N + n) * CHUNK + (m & 255)] = (f16)v;
                }
            }
        }
}

// x_ = silu(A @ Win^T + bin); emit qq (scaled), qk, lq row-major; lk chunk-transposed.
__global__ void __launch_bounds__(256)
k_gemm_in(const f16* __restrict__ A, const f16* __restrict__ W,
          const float* __restrict__ bias,
          const float* __restrict__ gqq, const float* __restrict__ bqq,
          const float* __restrict__ gqk, const float* __restrict__ bqk,
          const float* __restrict__ glq, const float* __restrict__ blq,
          const float* __restrict__ glk, const float* __restrict__ blk,
          f16* __restrict__ qq, f16* __restrict__ qk,
          f16* __restrict__ lq, f16* __restrict__ lkT, int K) {
    __shared__ f16 ldsA[TILE_F16], ldsB[TILE_F16];
    const int N  = HEAD_D;
    int       m0 = blockIdx.x * 128;
    v8f acc[2][4] = {};
    gemm_Kloop(A + (size_t)m0 * K, K, W, K, K, ldsA, ldsB, acc);

    int lane = threadIdx.x & 31, w = threadIdx.x >> 5;
    int wr = (w & 3) * 32, wc = (w >> 2) * 64;
#pragma unroll
    for (int r = 0; r < 2; ++r)
#pragma unroll
        for (int c = 0; c < 4; ++c) {
            int n = wc + c * 16 + (lane & 15);
#pragma unroll
            for (int j = 0; j < 8; ++j) {
                int    m  = m0 + wr + r * 16 + (lane >> 4) * 8 + j;
                float  x_ = silu_f(acc[r][c][j] + bias[n]);
                size_t o  = (size_t)m * N + n;
                qq[o] = (f16)((x_ * gqq[n] + bqq[n]) * INV_SCALE);
                qk[o] = (f16)(x_ * gqk[n] + bqk[n]);
                lq[o] = (f16)(x_ * glq[n] + blq[n]);
                lkT[((size_t)(m >> 8) * HEAD_D + n) * CHUNK + (m & 255)]
                    = (f16)(x_ * glk[n] + blk[n]);
            }
        }
}

// attn = relu(mask(qq @ qk^T))^2 per chunk.  grid=(2,2,NCHUNKS_TOT)
__global__ void __launch_bounds__(256)
k_qk(const f16* __restrict__ qq, const f16* __restrict__ qk, f16* __restrict__ attn) {
    __shared__ f16 ldsA[TILE_F16], ldsB[TILE_F16];
    int  chunk = blockIdx.z;
    int  m0 = blockIdx.x * 128, n0 = blockIdx.y * 128;
    f16* dst = attn + (size_t)chunk * CHUNK * CHUNK;

    if (n0 > m0 + 127) {  // tile fully above diagonal
        for (int i = threadIdx.x; i < 128 * 128; i += 256) {
            int m = m0 + i / 128, n = n0 + (i & 127);
            dst[(size_t)m * CHUNK + n] = (f16)0.f;
        }
        return;
    }

    v8f acc[2][4] = {};
    gemm_Kloop(qq + ((size_t)chunk * CHUNK + m0) * HEAD_D, HEAD_D,
               qk + ((size_t)chunk * CHUNK + n0) * HEAD_D, HEAD_D,
               HEAD_D, ldsA, ldsB, acc);

    int lane = threadIdx.x & 31, w = threadIdx.x >> 5;
    int wr = (w & 3) * 32, wc = (w >> 2) * 64;
#pragma unroll
    for (int r = 0; r < 2; ++r)
#pragma unroll
        for (int c = 0; c < 4; ++c) {
            int n = n0 + wc + c * 16 + (lane & 15);
#pragma unroll
            for (int j = 0; j < 8; ++j) {
                int   m = m0 + wr + r * 16 + (lane >> 4) * 8 + j;
                float v = acc[r][c][j];
                v = (n > m || v <= 0.f) ? 0.f : v * v;
                dst[(size_t)m * CHUNK + n] = (f16)v;
            }
        }
}

// kv_chunk = lin_k^T @ V per chunk (f32).  A=lkT[chunk]:[H,C], B=Vt[chunk]:[E,C].
__global__ void __launch_bounds__(256)
k_kv(const f16* __restrict__ lkT, const f16* __restrict__ Vt, float* __restrict__ kvc) {
    __shared__ f16 ldsA[TILE_F16], ldsB[TILE_F16];
    int chunk = blockIdx.z;
    int n0    = blockIdx.x * 128;
    v8f acc[2][4] = {};
    gemm_Kloop(lkT + (size_t)chunk * HEAD_D * CHUNK, CHUNK,
               Vt + ((size_t)chunk * EXP + n0) * CHUNK, CHUNK,
               CHUNK, ldsA, ldsB, acc);

    float* dst  = kvc + (size_t)chunk * HEAD_D * EXP;
    int    lane = threadIdx.x & 31, w = threadIdx.x >> 5;
    int    wr = (w & 3) * 32, wc = (w >> 2) * 64;
#pragma unroll
    for (int r = 0; r < 2; ++r)
#pragma unroll
        for (int c = 0; c < 4; ++c) {
            int n = n0 + wc + c * 16 + (lane & 15);
#pragma unroll
            for (int j = 0; j < 8; ++j) {
                int m = wr + r * 16 + (lane >> 4) * 8 + j;
                dst[(size_t)m * EXP + n] = acc[r][c][j];
            }
        }
}

// Exclusive prefix-sum over chunks; emit f16 transposed: kvexcT[chunk][E][H].
__global__ void k_scan(const float* __restrict__ kvc, f16* __restrict__ kvexcT) {
    size_t i = (size_t)blockIdx.x * blockDim.x + threadIdx.x;
    const size_t per_b = (size_t)HEAD_D * EXP;
    if (i >= (size_t)NB * per_b) return;
    size_t b = i / per_b, he = i % per_b;
    int    h = (int)(he / EXP), e = (int)(he % EXP);
    const float* src = kvc + b * NCHUNK * per_b + he;
    float acc = 0.f;
#pragma unroll
    for (int g = 0; g < NCHUNK; ++g) {
        kvexcT[(((size_t)b * NCHUNK + g) * EXP + e) * HEAD_D + h] = (f16)acc;
        acc += src[(size_t)g * per_b];
    }
}

// y = (attn @ V + lin_q @ kv_exc) * gate.  grid=(EXP/128, 2, NCHUNKS_TOT)
__global__ void __launch_bounds__(256)
k_attn(const f16* __restrict__ attn, const f16* __restrict__ Vt,
       const f16* __restrict__ lq, const f16* __restrict__ kvexcT,
       const f16* __restrict__ gate, f16* __restrict__ y) {
    __shared__ f16 ldsA0[TILE_F16], ldsB0[TILE_F16], ldsA1[TILE_F16], ldsB1[TILE_F16];
    int chunk = blockIdx.z;
    int n0    = blockIdx.x * 128;
    int m0    = blockIdx.y * 128;
    v8f acc[2][4] = {};
    // pass 1: local quadratic attention (K = CHUNK)
    gemm_Kloop_tdm(attn + (size_t)chunk * CHUNK * CHUNK + (size_t)m0 * CHUNK, CHUNK,
                   Vt + ((size_t)chunk * EXP + n0) * CHUNK, CHUNK, CHUNK,
                   ldsA0, ldsB0, ldsA1, ldsB1, acc);
    // pass 2: global linear attention (K = HEAD_D), same accumulators
    gemm_Kloop_tdm(lq + ((size_t)chunk * CHUNK + m0) * HEAD_D, HEAD_D,
                   kvexcT + ((size_t)chunk * EXP + n0) * HEAD_D, HEAD_D, HEAD_D,
                   ldsA0, ldsB0, ldsA1, ldsB1, acc);

    int lane = threadIdx.x & 31, w = threadIdx.x >> 5;
    int wr = (w & 3) * 32, wc = (w >> 2) * 64;
#pragma unroll
    for (int r = 0; r < 2; ++r)
#pragma unroll
        for (int c = 0; c < 4; ++c) {
            int n = n0 + wc + c * 16 + (lane & 15);
#pragma unroll
            for (int j = 0; j < 8; ++j) {
                int    m   = wr + r * 16 + (lane >> 4) * 8 + j;
                size_t row = (size_t)chunk * CHUNK + m0 + m;
                float  g   = (float)gate[row * EXP + n];
                y[row * EXP + n] = (f16)(acc[r][c][j] * g);
            }
        }
}

// out = y @ Wout^T + bout (fp32 output).  W:[DIM,EXP].  grid=(ROWS/128, DIM/128)
__global__ void __launch_bounds__(256)
k_out(const f16* __restrict__ y, const f16* __restrict__ W,
      const float* __restrict__ bout, float* __restrict__ out) {
    __shared__ f16 ldsA0[TILE_F16], ldsB0[TILE_F16], ldsA1[TILE_F16], ldsB1[TILE_F16];
    int m0 = blockIdx.x * 128, n0 = blockIdx.y * 128;
    v8f acc[2][4] = {};
    gemm_Kloop_tdm(y + (size_t)m0 * EXP, EXP, W + (size_t)n0 * EXP, EXP, EXP,
                   ldsA0, ldsB0, ldsA1, ldsB1, acc);

    int lane = threadIdx.x & 31, w = threadIdx.x >> 5;
    int wr = (w & 3) * 32, wc = (w >> 2) * 64;
#pragma unroll
    for (int r = 0; r < 2; ++r)
#pragma unroll
        for (int c = 0; c < 4; ++c) {
            int   n  = n0 + wc + c * 16 + (lane & 15);
            float bn = bout[n];
#pragma unroll
            for (int j = 0; j < 8; ++j) {
                int m = m0 + wr + r * 16 + (lane >> 4) * 8 + j;
                out[(size_t)m * DIM + n] = acc[r][c][j] + bn;
            }
        }
}

// ---------------------------------------------------------------------------
// Host launch
// ---------------------------------------------------------------------------
extern "C" void kernel_launch(void* const* d_in, const int* in_sizes, int n_in,
                              void* d_out, int out_size, void* d_ws, size_t ws_size,
                              hipStream_t stream) {
    const float* x    = (const float*)d_in[0];
    const float* Wv   = (const float*)d_in[1];
    const float* bv   = (const float*)d_in[2];
    const float* Wg   = (const float*)d_in[3];
    const float* bg   = (const float*)d_in[4];
    const float* Win  = (const float*)d_in[5];
    const float* bin_ = (const float*)d_in[6];
    const float* Wout = (const float*)d_in[7];
    const float* bout = (const float*)d_in[8];
    const float* gqq  = (const float*)d_in[9];
    const float* bqq  = (const float*)d_in[10];
    const float* gqk  = (const float*)d_in[11];
    const float* bqk  = (const float*)d_in[12];
    const float* glq  = (const float*)d_in[13];
    const float* blq  = (const float*)d_in[14];
    const float* glk  = (const float*)d_in[15];
    const float* blk  = (const float*)d_in[16];
    float* out = (float*)d_out;

    // --- workspace carve ---
    char* p = (char*)d_ws;
    auto alloc = [&](size_t bytes) -> char* {
        char* r = p;
        p += (bytes + 255) & ~(size_t)255;
        return r;
    };
    f16*   xh     = (f16*)alloc((size_t)ROWS * DIM * 2);
    f16*   WvT    = (f16*)alloc((size_t)DIM * EXP * 2);      // [E][D]
    f16*   WgT    = (f16*)alloc((size_t)DIM * EXP * 2);      // [E][D]
    f16*   WinT   = (f16*)alloc((size_t)DIM * HEAD_D * 2);   // [H][D]
    f16*   WoutT  = (f16*)alloc((size_t)EXP * DIM * 2);      // [D][E]
    f16*   Vt     = (f16*)alloc((size_t)ROWS * EXP * 2);     // [chunk][E][C]
    f16*   Gh     = (f16*)alloc((size_t)ROWS * EXP * 2);     // row-major
    f16*   qq     = (f16*)alloc((size_t)ROWS * HEAD_D * 2);
    f16*   qk     = (f16*)alloc((size_t)ROWS * HEAD_D * 2);
    f16*   lq     = (f16*)alloc((size_t)ROWS * HEAD_D * 2);
    f16*   lkT    = (f16*)alloc((size_t)ROWS * HEAD_D * 2);  // [chunk][H][C]
    f16*   attn   = (f16*)alloc((size_t)NCHUNKS_TOT * CHUNK * CHUNK * 2);
    float* kvc    = (float*)alloc((size_t)NCHUNKS_TOT * HEAD_D * EXP * 4);
    f16*   kvexcT = (f16*)alloc((size_t)NCHUNKS_TOT * HEAD_D * EXP * 2); // [chunk][E][H]
    f16*   ymid   = (f16*)alloc((size_t)ROWS * EXP * 2);

    // --- 1. conversions (x plain; weights transpose-converted) ---
    {
        int n = ROWS * DIM;
        k_convert<<<4096, 256, 0, stream>>>(x, xh, n);
    }
    k_convT<<<4096, 256, 0, stream>>>(Wv,   WvT,   DIM, EXP);
    k_convT<<<4096, 256, 0, stream>>>(Wg,   WgT,   DIM, EXP);
    k_convT<<<512,  256, 0, stream>>>(Win,  WinT,  DIM, HEAD_D);
    k_convT<<<4096, 256, 0, stream>>>(Wout, WoutT, EXP, DIM);

    // --- 2. projections: V (chunk-transposed) and Gate (row-major) ---
    dim3 gVG(ROWS / 128, EXP / 128);
    k_gemm_silu<<<gVG, 256, 0, stream>>>(xh, WvT, bv, Vt, EXP, DIM, 1);
    k_gemm_silu<<<gVG, 256, 0, stream>>>(xh, WgT, bg, Gh, EXP, DIM, 0);

    // --- 3. head projection -> quad/lin q,k ---
    k_gemm_in<<<dim3(ROWS / 128, 1), 256, 0, stream>>>(
        xh, WinT, bin_, gqq, bqq, gqk, bqk, glq, blq, glk, blk,
        qq, qk, lq, lkT, DIM);

    // --- 4. per-chunk masked relu^2 attention scores ---
    k_qk<<<dim3(2, 2, NCHUNKS_TOT), 256, 0, stream>>>(qq, qk, attn);

    // --- 5. per-chunk k^T v ---
    k_kv<<<dim3(EXP / 128, 1, NCHUNKS_TOT), 256, 0, stream>>>(lkT, Vt, kvc);

    // --- 6. exclusive chunk scan (emits transposed f16) ---
    {
        size_t n = (size_t)NB * HEAD_D * EXP;
        k_scan<<<(unsigned)((n + 255) / 256), 256, 0, stream>>>(kvc, kvexcT);
    }

    // --- 7. quadratic + linear attention, gated ---
    k_attn<<<dim3(EXP / 128, 2, NCHUNKS_TOT), 256, 0, stream>>>(
        attn, Vt, lq, kvexcT, Gh, ymid);

    // --- 8. output projection ---
    k_out<<<dim3(ROWS / 128, DIM / 128), 256, 0, stream>>>(ymid, WoutT, bout, out);
}
